// Attention_56418690401007
// MI455X (gfx1250) — compile-verified, matrix operand order checked by reading, not run on hardware
//
#include <hip/hip_runtime.h>

// ---------------------------------------------------------------------------
// CDNA5 (gfx1250) causal multi-head attention block:
//   prep : x, W_attn^T, W_proj^T converted to bf16 in workspace (once)
//   qkv  = x @ W_attn + b_attn   (async-LDS double-buffered bf16 WMMA GEMM)
//   y    = flash_attention(q,k,v) (bf16 WMMA, online softmax)
//   out  = y @ W_proj + b_proj   (same GEMM, fp32 out)
// B=4, T=2048, C=1024, H=16, hd=64. Wave32, V_WMMA_F32_16X16X32_BF16,
// GLOBAL_LOAD_ASYNC_TO_LDS_B128 + S_WAIT_ASYNCCNT for tile staging.
// ---------------------------------------------------------------------------

typedef __attribute__((ext_vector_type(16))) __bf16 v16bf;
typedef __attribute__((ext_vector_type(8)))  float  v8f;

union Frag16 { v16bf v; unsigned u[8]; };

__device__ __forceinline__ unsigned short f32_to_bf16(float f) {
  union { float f; unsigned u; } x; x.f = f;
  unsigned r = x.u + 0x7FFFu + ((x.u >> 16) & 1u);   // round-to-nearest-even
  return (unsigned short)(r >> 16);
}

__device__ __forceinline__ unsigned pack_bf16x2(float lo, float hi) {
  return (unsigned)f32_to_bf16(lo) | ((unsigned)f32_to_bf16(hi) << 16);
}

__device__ __forceinline__ v8f wmma_bf16(const Frag16& a, const Frag16& b, v8f c) {
  return __builtin_amdgcn_wmma_f32_16x16x32_bf16(
      /*neg_a=*/false, a.v, /*neg_b=*/false, b.v,
      /*c_mod=*/(short)0, c, /*reuse_a=*/false, /*reuse_b=*/false);
}

// A-matrix 16x32 bf16 VGPR layout (ISA 7.12.2):
//   lane L: M = L%16, half = L/16
//   dword v: k = (v>=4?16:0) + half*8 + (v&3)*2  (two packed bf16: k, k+1)
__device__ __forceinline__ int frag_k(int v, int lh) {
  return ((v >> 2) << 4) + (lh << 3) + ((v & 3) << 1);
}

__device__ __forceinline__ void store_out(float* p, float v) { *p = v; }
__device__ __forceinline__ void store_out(unsigned short* p, float v) { *p = f32_to_bf16(v); }

// Raw 32-bit LDS offset of a __shared__ object (generic shared addr low bits).
__device__ __forceinline__ unsigned lds_off(const void* p) {
  return (unsigned)(unsigned long long)p;
}

// Async global -> LDS copy of 16 bytes per lane (ASYNCcnt-tracked).
__device__ __forceinline__ void async_load_b128(unsigned lds, const void* gaddr) {
  asm volatile("global_load_async_to_lds_b128 %0, %1, off"
               :: "v"(lds), "v"(gaddr) : "memory");
}
__device__ __forceinline__ void wait_asynccnt0() {
  asm volatile("s_wait_asynccnt 0" ::: "memory");
}

// ---------------------------------------------------------------------------
// Elementwise fp32 -> bf16 (vectorized, one float4 per thread).
// ---------------------------------------------------------------------------
__global__ __launch_bounds__(256)
void f32_to_bf16_kernel(const float* __restrict__ src,
                        unsigned short* __restrict__ dst, int n4) {
  const int i = blockIdx.x * 256 + threadIdx.x;
  if (i < n4) {
    const float4 v = reinterpret_cast<const float4*>(src)[i];
    uint2 o;
    o.x = pack_bf16x2(v.x, v.y);
    o.y = pack_bf16x2(v.z, v.w);
    reinterpret_cast<uint2*>(dst)[i] = o;
  }
}

// ---------------------------------------------------------------------------
// W[K][N] fp32  ->  Wt[N][K] bf16  (32x32 tiles through LDS).
// ---------------------------------------------------------------------------
__global__ __launch_bounds__(256)
void transpose_f32_to_bf16_kernel(const float* __restrict__ W,
                                  unsigned short* __restrict__ Wt,
                                  int K, int N) {
  __shared__ float tile[32][33];
  const int k0 = blockIdx.y * 32;
  const int n0 = blockIdx.x * 32;
  const int c  = threadIdx.x & 31;
  const int r0 = threadIdx.x >> 5;   // 0..7
#pragma unroll
  for (int i = 0; i < 4; ++i) {
    const int r = r0 + i * 8;
    tile[r][c] = W[(size_t)(k0 + r) * N + n0 + c];
  }
  __syncthreads();
#pragma unroll
  for (int i = 0; i < 4; ++i) {
    const int r = r0 + i * 8;        // output row along n
    Wt[(size_t)(n0 + r) * K + k0 + c] = f32_to_bf16(tile[c][r]);
  }
}

// ---------------------------------------------------------------------------
// GEMM: C[M,N] = A[M,K](bf16,row-major) @ Bt[N,K](bf16,pre-transposed) + bias.
// Workgroup: 256 threads (8 waves), tile 128(M) x 64(N), K step 32,
// double-buffered LDS filled with GLOBAL_LOAD_ASYNC_TO_LDS_B128.
// Wave (waveM 0..3, waveN 0..1) owns a 32x32 block = 2x2 WMMA tiles.
// LDS row pitch 40 elems = 80 B: 16B-aligned for b128 async, 20-dword pitch
// (gcd(20,64)=4, first 16 multiples distinct mod 64 -> conflict-free frags).
// ---------------------------------------------------------------------------
template <typename OT>
__global__ __launch_bounds__(256, 2)
void gemm_bias_async_wmma_kernel(const unsigned short* __restrict__ A,
                                 const unsigned short* __restrict__ Bt,
                                 const float* __restrict__ bias,
                                 OT* __restrict__ Cc, int M, int N, int K) {
  constexpr int BM = 128, BN = 64, BK = 32, ST = 40;
  __shared__ unsigned short Asl[2][BM * ST];   // [stage][m][k]
  __shared__ unsigned short Bsl[2][BN * ST];   // [stage][n][k]

  const int tid  = threadIdx.x;
  const int wave = tid >> 5;
  const int lane = tid & 31;
  const int lm = lane & 15, lh = lane >> 4;
  const int waveM = wave >> 1, waveN = wave & 1;
  const long long blockM = (long long)blockIdx.y * BM;
  const long long blockN = (long long)blockIdx.x * BN;

  // Issue one stage's tiles: A = 512 b128 transfers, B = 256 -> 3 per thread.
  auto issue = [&](int k0, int stage) {
#pragma unroll
    for (int i = 0; i < 2; ++i) {
      const int f = tid + i * 256;           // 0..511
      const int r = f >> 2, j = f & 3;       // row, 16B chunk
      async_load_b128(lds_off(&Asl[stage][r * ST + j * 8]),
                      A + (blockM + r) * (long long)K + k0 + j * 8);
    }
    {
      const int n = tid >> 2, j = tid & 3;
      async_load_b128(lds_off(&Bsl[stage][n * ST + j * 8]),
                      Bt + (blockN + n) * (long long)K + k0 + j * 8);
    }
  };

  v8f zero = {};
  v8f acc[2][2];
  for (int i = 0; i < 2; ++i)
    for (int j = 0; j < 2; ++j) acc[i][j] = zero;

  const int nIter = K / BK;
  issue(0, 0);

  for (int it = 0; it < nIter; ++it) {
    const int stage = it & 1;
    wait_asynccnt0();       // this wave's async transfers landed in LDS
    __syncthreads();        // ...and everyone else's
    if (it + 1 < nIter) issue((it + 1) * BK, stage ^ 1);

    const unsigned short* Ab = Asl[stage];
    const unsigned short* Bb = Bsl[stage];

    Frag16 af[2], bf[2];
#pragma unroll
    for (int im = 0; im < 2; ++im) {
      const int mrow = waveM * 32 + im * 16 + lm;
#pragma unroll
      for (int v = 0; v < 8; ++v)
        af[im].u[v] = *reinterpret_cast<const unsigned*>(&Ab[mrow * ST + frag_k(v, lh)]);
    }
#pragma unroll
    for (int in = 0; in < 2; ++in) {
      const int ncol = waveN * 32 + in * 16 + lm;
#pragma unroll
      for (int v = 0; v < 8; ++v)
        bf[in].u[v] = *reinterpret_cast<const unsigned*>(&Bb[ncol * ST + frag_k(v, lh)]);
    }
#pragma unroll
    for (int im = 0; im < 2; ++im)
#pragma unroll
      for (int in = 0; in < 2; ++in)
        acc[im][in] = wmma_bf16(af[im], bf[in], acc[im][in]);

    __syncthreads();        // LDS reads done before this buffer is refilled
  }

  // Epilogue: C-layout (vgpr r -> M = r + 8*lh, N = lane%16) + bias.
#pragma unroll
  for (int im = 0; im < 2; ++im) {
    const long long mb = blockM + waveM * 32 + im * 16 + (lh << 3);
#pragma unroll
    for (int in = 0; in < 2; ++in) {
      const long long nn = blockN + waveN * 32 + in * 16 + lm;
      const float bv = bias[nn];
#pragma unroll
      for (int r = 0; r < 8; ++r)
        store_out(Cc + (mb + r) * N + nn, acc[im][in][r] + bv);
    }
  }
}

// ---------------------------------------------------------------------------
// Flash attention (causal): grid (T/64, B*H), 128 threads = 4 waves.
// Wave w owns 16 query rows; streams 32-key blocks with online softmax.
// qkv: bf16 [B*T, 3C] (q | k | v per row), y: bf16 [B*T, C].
// ---------------------------------------------------------------------------
__global__ __launch_bounds__(128, 4)
void flash_attn_wmma_kernel(const unsigned short* __restrict__ qkv,
                            unsigned short* __restrict__ y) {
  constexpr int T = 2048, HD = 64, C = 1024, C3 = 3072;
  constexpr int PS = 36;
  __shared__ unsigned short Ps[4][16 * PS];  // per-wave P staging (C-layout -> A-layout)

  const int w = threadIdx.x >> 5;
  const int lane = threadIdx.x & 31;
  const int lm = lane & 15, lh = lane >> 4;
  const int b = blockIdx.y >> 4;   // / H
  const int h = blockIdx.y & 15;   // % H
  const int qBase = blockIdx.x * 64 + w * 16;
  const long long rowsBase = (long long)b * T;

  // Q fragments (d-halves 0..31, 32..63), A-layout straight from global bf16.
  Frag16 qf[2];
  {
    const long long qoff = (rowsBase + qBase + lm) * C3 + h * HD;
#pragma unroll
    for (int half = 0; half < 2; ++half)
#pragma unroll
      for (int v = 0; v < 8; ++v)
        qf[half].u[v] =
            *reinterpret_cast<const unsigned*>(qkv + qoff + half * 32 + frag_k(v, lh));
  }

  v8f zero = {};
  v8f accO[4];
  float mrun[8], lrun[8];
#pragma unroll
  for (int t = 0; t < 4; ++t) accO[t] = zero;
#pragma unroll
  for (int r = 0; r < 8; ++r) { mrun[r] = -__builtin_inff(); lrun[r] = 0.0f; }

  const float scale = 0.125f;  // 1/sqrt(64)

  for (int kBase = 0; kBase <= qBase + 15; kBase += 32) {
    // --- S = Q K^T for two 16-key sub-tiles (K over hd = 2 x 32) ---
    v8f s[2];
#pragma unroll
    for (int j = 0; j < 2; ++j) {
      v8f sj = zero;
      const long long koff = (rowsBase + kBase + j * 16 + lm) * C3 + C + h * HD;
#pragma unroll
      for (int half = 0; half < 2; ++half) {
        Frag16 kf;
#pragma unroll
        for (int v = 0; v < 8; ++v)
          kf.u[v] =
              *reinterpret_cast<const unsigned*>(qkv + koff + half * 32 + frag_k(v, lh));
        sj = wmma_bf16(qf[half], kf, sj);
      }
      s[j] = sj;
    }

    // --- scale + causal mask (per-lane select, no divergence) ---
#pragma unroll
    for (int j = 0; j < 2; ++j)
#pragma unroll
      for (int r = 0; r < 8; ++r) {
        const int qrow = qBase + r + (lh << 3);
        const int key  = kBase + j * 16 + lm;
        const float val = s[j][r] * scale;
        s[j][r] = (key <= qrow) ? val : -__builtin_inff();
      }

    // --- online softmax: row stats across 16 lanes of each half ---
#pragma unroll
    for (int r = 0; r < 8; ++r) {
      float mx = fmaxf(s[0][r], s[1][r]);
#pragma unroll
      for (int d = 8; d >= 1; d >>= 1) mx = fmaxf(mx, __shfl_xor(mx, d, 32));
      const float mnew  = fmaxf(mrun[r], mx);
      const float alpha = __expf(mrun[r] - mnew);
      mrun[r] = mnew;
      const float p0 = __expf(s[0][r] - mnew);
      const float p1 = __expf(s[1][r] - mnew);
      s[0][r] = p0; s[1][r] = p1;
      float rs = p0 + p1;
#pragma unroll
      for (int d = 8; d >= 1; d >>= 1) rs += __shfl_xor(rs, d, 32);
      lrun[r] = lrun[r] * alpha + rs;
#pragma unroll
      for (int t = 0; t < 4; ++t) accO[t][r] *= alpha;
    }

    // --- P: C-layout -> LDS (wave-private, DS ops are wave-in-order) ---
#pragma unroll
    for (int j = 0; j < 2; ++j)
#pragma unroll
      for (int r = 0; r < 8; ++r)
        Ps[w][(r + (lh << 3)) * PS + j * 16 + lm] = f32_to_bf16(s[j][r]);
    asm volatile("" ::: "memory");  // keep ds_store before ds_load

    Frag16 pf;  // P as 16x32 A-fragment (K = 32 keys)
#pragma unroll
    for (int v = 0; v < 8; ++v)
      pf.u[v] = *reinterpret_cast<const unsigned*>(&Ps[w][lm * PS + frag_k(v, lh)]);

    // --- accO += P @ V (4 tiles over hd) ---
#pragma unroll
    for (int t = 0; t < 4; ++t) {
      Frag16 vf;
#pragma unroll
      for (int v = 0; v < 8; ++v) {
        const int kk = frag_k(v, lh);
        const long long vrow =
            (rowsBase + kBase + kk) * C3 + 2 * C + h * HD + t * 16 + lm;
        const unsigned lo = qkv[vrow];
        const unsigned hi = qkv[vrow + C3];
        vf.u[v] = lo | (hi << 16);
      }
      accO[t] = wmma_bf16(pf, vf, accO[t]);
    }
  }

  // --- normalize + store y as bf16 [B*T, C] (heads back in natural order) ---
  float inv[8];
#pragma unroll
  for (int r = 0; r < 8; ++r) inv[r] = 1.0f / lrun[r];
#pragma unroll
  for (int t = 0; t < 4; ++t)
#pragma unroll
    for (int r = 0; r < 8; ++r) {
      const long long row = rowsBase + qBase + r + (lh << 3);
      y[row * C + h * HD + t * 16 + lm] = f32_to_bf16(accO[t][r] * inv[r]);
    }
}

// ---------------------------------------------------------------------------
extern "C" void kernel_launch(void* const* d_in, const int* in_sizes, int n_in,
                              void* d_out, int out_size, void* d_ws, size_t ws_size,
                              hipStream_t stream) {
  (void)in_sizes; (void)n_in; (void)out_size; (void)ws_size;
  constexpr int B = 4, T = 2048, C = 1024, H = 16;
  constexpr int M = B * T;

  const float* x      = (const float*)d_in[0];
  const float* W_attn = (const float*)d_in[1];
  const float* b_attn = (const float*)d_in[2];
  const float* W_proj = (const float*)d_in[3];
  const float* b_proj = (const float*)d_in[4];
  float* out = (float*)d_out;

  // Workspace (bf16 elems): qkv | y | x | W_attn^T | W_proj^T  (~92 MB)
  unsigned short* qkvB = (unsigned short*)d_ws;
  unsigned short* yB   = qkvB + (size_t)M * 3 * C;
  unsigned short* xB   = yB   + (size_t)M * C;
  unsigned short* WaT  = xB   + (size_t)M * C;      // [3C][C]
  unsigned short* WpT  = WaT  + (size_t)3 * C * C;  // [C][C]

  // 0) one-time bf16 conversion / weight transpose
  const int n4x = M * C / 4;
  f32_to_bf16_kernel<<<dim3((n4x + 255) / 256), dim3(256), 0, stream>>>(x, xB, n4x);
  transpose_f32_to_bf16_kernel<<<dim3(3 * C / 32, C / 32), dim3(256), 0, stream>>>(
      W_attn, WaT, C, 3 * C);
  transpose_f32_to_bf16_kernel<<<dim3(C / 32, C / 32), dim3(256), 0, stream>>>(
      W_proj, WpT, C, C);

  // 1) qkv = x @ W_attn + b_attn  (bf16 out)
  gemm_bias_async_wmma_kernel<unsigned short>
      <<<dim3(3 * C / 64, M / 128), dim3(256), 0, stream>>>(
          xB, WaT, b_attn, qkvB, M, 3 * C, C);

  // 2) y = causal flash attention over qkv
  flash_attn_wmma_kernel<<<dim3(T / 64, B * H), dim3(128), 0, stream>>>(qkvB, yB);

  // 3) out = y @ W_proj + b_proj  (fp32 out)
  gemm_bias_async_wmma_kernel<float>
      <<<dim3(C / 64, M / 128), dim3(256), 0, stream>>>(
          yB, WpT, b_proj, out, M, C, C);
}
